// Head_49503793054210
// MI455X (gfx1250) — matrix operand order repaired and checked by reading, not compiled, not run on hardware
//
#include <hip/hip_runtime.h>
#include <hip/hip_bf16.h>

// MI455X / gfx1250, wave32. bf16 WMMA (16x16x32) + fp32 accum.
// Workspace (needs 48 MB): kb [B][T][H] | qb [B][T][H] | vbT [B][H][T], bf16.

typedef __attribute__((ext_vector_type(16))) __bf16 v16bf;
typedef __attribute__((ext_vector_type(8)))  float  v8f;

#define WMMA_BF16(a, b, c) \
  __builtin_amdgcn_wmma_f32_16x16x32_bf16(false, (a), false, (b), (short)0, (c), false, false)

static constexpr int kB = 256;   // batch
static constexpr int kT = 256;   // sequence
static constexpr int kC = 1024;  // model dim
static constexpr int kH = 128;   // head dim

// --- fragment loaders: ISA 7.12.2 layouts collapse to contiguous 16B runs ---
__device__ __forceinline__ v16bf load_afrag(const __bf16* row, int khalf) {
  union { v16bf v; float4 f[2]; } u;
  u.f[0] = *(const float4*)(row + khalf * 8);
  u.f[1] = *(const float4*)(row + 16 + khalf * 8);
  return u.v;
}
__device__ __forceinline__ v16bf load_bfrag(const __bf16* col_k, int khalf) {
  union { v16bf v; float4 f[2]; } u;
  u.f[0] = *(const float4*)(col_k + khalf * 8);
  u.f[1] = *(const float4*)(col_k + 16 + khalf * 8);
  return u.v;
}

__device__ __forceinline__ unsigned lds_byte_off(const void* p) {
  return (unsigned)(size_t)p;  // low 32 bits of generic LDS addr = LDS offset
}
// CDNA5 async DMA, tracked by ASYNCcnt
__device__ __forceinline__ void async_load_b128(unsigned lds_dst, const void* gsrc) {
  asm volatile("global_load_async_to_lds_b128 %0, %1, off"
               :: "v"(lds_dst), "v"((unsigned long long)(size_t)gsrc) : "memory");
}
__device__ __forceinline__ void async_store_b128(void* gdst, unsigned lds_src) {
  asm volatile("global_store_async_from_lds_b128 %0, %1, off"
               :: "v"((unsigned long long)(size_t)gdst), "v"(lds_src) : "memory");
}
__device__ __forceinline__ void wait_async0() {
  asm volatile("s_wait_asynccnt 0x0" ::: "memory");
}
__device__ __forceinline__ unsigned pack_bf16(float lo, float hi) {
  union { __bf16 h[2]; unsigned u; } p;
  p.h[0] = (__bf16)lo; p.h[1] = (__bf16)hi;
  return p.u;
}

// ---------------------------------------------------------------------------
// Kernel 1: D = X @ W (bf16 WMMA, fp32 accum). 256 thr = 8 waves, 128x128 tile.
// Epilogue bounces C through LDS and streams it out with async b128 stores.
// TRANSPOSED=true stores D as [B][H][T] (for V).
// ---------------------------------------------------------------------------
template <bool TRANSPOSED>
__global__ __launch_bounds__(256) void qkv_bf16_kernel(
    const float* __restrict__ x, const float* __restrict__ W0,
    const float* __restrict__ W1, __bf16* __restrict__ D0, __bf16* __restrict__ D1)
{
  __shared__ __align__(16) __bf16 ldsA[128][32];   // X tile (M x K)
  __shared__ __align__(16) __bf16 ldsBT[128][32];  // W tile (N x K), transposed
  __shared__ __align__(16) __bf16 ldsC[128][128];  // C bounce tile

  const int tid   = threadIdx.x;
  const int wave  = tid >> 5;
  const int lane  = tid & 31;
  const int col   = lane & 15;
  const int khalf = lane >> 4;
  const int rowBase = blockIdx.x * 128;

  const float* W = (blockIdx.y == 0) ? W0 : W1;
  __bf16*      D = (blockIdx.y == 0) ? D0 : D1;

  v8f vzero = {};
  v8f acc[8];
  #pragma unroll
  for (int i = 0; i < 8; ++i) acc[i] = vzero;

  for (int k0 = 0; k0 < kC; k0 += 32) {
    __syncthreads();
    { // stage X tile 128x32: fp32 -> bf16, vector loads + b128 LDS stores
      int r  = tid >> 1;
      int c0 = (tid & 1) * 16;
      const float4* src = (const float4*)(x + (size_t)(rowBase + r) * kC + k0 + c0);
      union { __bf16 h[16]; float4 f[2]; } t;
      #pragma unroll
      for (int q4 = 0; q4 < 4; ++q4) {
        float4 f = src[q4];
        t.h[q4 * 4 + 0] = (__bf16)f.x; t.h[q4 * 4 + 1] = (__bf16)f.y;
        t.h[q4 * 4 + 2] = (__bf16)f.z; t.h[q4 * 4 + 3] = (__bf16)f.w;
      }
      *(float4*)&ldsA[r][c0]     = t.f[0];
      *(float4*)&ldsA[r][c0 + 8] = t.f[1];
      if (k0 + 32 < kC) __builtin_prefetch(src + 8, 0, 0);   // next K tile of X
    }
    { // stage W tile 32x128 transposed -> ldsBT[n][k], row-pairs -> b32 stores
      int r2 = (tid >> 4) * 2;       // k rows r2, r2+1
      int c0 = (tid & 15) * 8;       // 8 n columns
      const float4* s0p = (const float4*)(W + (size_t)(k0 + r2) * kH + c0);
      const float4* s1p = (const float4*)(W + (size_t)(k0 + r2 + 1) * kH + c0);
      float4 a0 = s0p[0], a1 = s0p[1];
      float4 b0 = s1p[0], b1 = s1p[1];
      float r0v[8] = {a0.x, a0.y, a0.z, a0.w, a1.x, a1.y, a1.z, a1.w};
      float r1v[8] = {b0.x, b0.y, b0.z, b0.w, b1.x, b1.y, b1.z, b1.w};
      #pragma unroll
      for (int n = 0; n < 8; ++n)
        *(unsigned*)&ldsBT[c0 + n][r2] = pack_bf16(r0v[n], r1v[n]);
      if (k0 + 32 < kC) __builtin_prefetch((const float*)s0p + 32 * kH, 0, 0);
    }
    __syncthreads();

    v16bf afrag = load_afrag(&ldsA[wave * 16 + col][0], khalf);
    #pragma unroll
    for (int nt = 0; nt < 8; ++nt) {
      v16bf bfrag = load_bfrag(&ldsBT[nt * 16 + col][0], khalf);
      acc[nt] = WMMA_BF16(afrag, bfrag, acc[nt]);
    }
  }

  // epilogue: C (VGPR r -> M = r + 8*khalf, N = col) -> LDS -> async b128 out
  if constexpr (TRANSPOSED) {
    #pragma unroll
    for (int nt = 0; nt < 8; ++nt)
      #pragma unroll
      for (int r2 = 0; r2 < 8; r2 += 2)   // ldsC[h][m], m-pairs contiguous
        *(unsigned*)&ldsC[nt * 16 + col][wave * 16 + 8 * khalf + r2] =
            pack_bf16(acc[nt][r2], acc[nt][r2 + 1]);
  } else {
    #pragma unroll
    for (int nt = 0; nt < 8; ++nt)
      #pragma unroll
      for (int r = 0; r < 8; ++r)
        ldsC[wave * 16 + 8 * khalf + r][nt * 16 + col] = (__bf16)acc[nt][r];
  }
  __syncthreads();
  {
    int row = tid >> 1;                 // output row (m or h)
    int c0  = (tid & 1) * 64;           // 64-element half-row
    unsigned lsrc = lds_byte_off(&ldsC[row][c0]);
    __bf16* dst;
    if constexpr (TRANSPOSED) {
      int bb = rowBase >> 8, tbase = rowBase & 255;
      dst = D + (size_t)bb * kH * kT + (size_t)row * kT + tbase + c0;
    } else {
      dst = D + (size_t)(rowBase + row) * kH + c0;
    }
    #pragma unroll
    for (int i = 0; i < 8; ++i)
      async_store_b128(dst + i * 8, lsrc + i * 16);
    wait_async0();
  }
}

// ---------------------------------------------------------------------------
// Kernel 2: causal attention, streaming softmax. grid=(B, T/128), 8 waves.
// Q / Vt blocks staged via GLOBAL_LOAD_ASYNC_TO_LDS_B128.
// ---------------------------------------------------------------------------
__global__ __launch_bounds__(256) void attn_kernel(
    const __bf16* __restrict__ kb, const __bf16* __restrict__ qb,
    const __bf16* __restrict__ vbT, float* __restrict__ out)
{
  __shared__ __align__(16) __bf16 q_blk[32][128];    // q rows  [s][h]
  __shared__ __align__(16) __bf16 v_blkT[128][32];   // v cols  [h][s]
  __shared__ __align__(16) __bf16 p_lds[8][16][32];  // per-wave P staging

  const int tid   = threadIdx.x;
  const int wave  = tid >> 5;
  const int lane  = tid & 31;
  const int col   = lane & 15;
  const int khalf = lane >> 4;
  const int b     = blockIdx.x;
  const int t0    = blockIdx.y * 128 + wave * 16;
  const float scale = 0.08838834764831845f;  // 1/sqrt(128)

  v16bf kf[4];
  {
    const __bf16* krow = kb + (size_t)(b * kT + t0 + col) * kH;
    #pragma unroll
    for (int c4 = 0; c4 < 4; ++c4) kf[c4] = load_afrag(krow + c4 * 32, khalf);
  }

  v8f vzero = {};
  float m[8], l[8];
  v8f acc[8];
  #pragma unroll
  for (int r = 0; r < 8; ++r) { m[r] = -1e30f; l[r] = 0.0f; }
  #pragma unroll
  for (int h = 0; h < 8; ++h) acc[h] = vzero;

  const unsigned q_base = lds_byte_off(&q_blk[0][0]);
  const unsigned v_base = lds_byte_off(&v_blkT[0][0]);

  for (int s0 = 0; s0 < kT; s0 += 32) {
    __syncthreads();
    { // async-stage 32 q rows (8 KB) and 32 vT cols (8 KB)
      const __bf16* qsrc = qb + (size_t)(b * kT + s0) * kH;
      const __bf16* vsrc = vbT + (size_t)b * kH * kT + s0;
      #pragma unroll
      for (int i = 0; i < 2; ++i) {
        int c = tid + i * 256;
        async_load_b128(q_base + c * 16, qsrc + c * 8);
        async_load_b128(v_base + c * 16, vsrc + (size_t)(c >> 2) * kT + (c & 3) * 8);
      }
      wait_async0();
    }
    __syncthreads();

    if (s0 <= t0 + 15) {  // causal block skip (uniform per wave)
      v8f S[2];
      #pragma unroll
      for (int nt = 0; nt < 2; ++nt) {
        S[nt] = vzero;
        #pragma unroll
        for (int c4 = 0; c4 < 4; ++c4) {
          v16bf bq = load_bfrag(&q_blk[nt * 16 + col][c4 * 32], khalf);
          S[nt] = WMMA_BF16(kf[c4], bq, S[nt]);
        }
      }
      #pragma unroll
      for (int nt = 0; nt < 2; ++nt)
        #pragma unroll
        for (int r = 0; r < 8; ++r) {
          int t = t0 + r + 8 * khalf;
          int s = s0 + nt * 16 + col;
          float v = S[nt][r] * scale;
          S[nt][r] = (s <= t) ? v : -1e30f;
        }
      float corr[8];
      #pragma unroll
      for (int r = 0; r < 8; ++r) {
        float rm = fmaxf(S[0][r], S[1][r]);
        #pragma unroll
        for (int off = 1; off < 16; off <<= 1)
          rm = fmaxf(rm, __shfl_xor(rm, off, 32));
        float mnew = fmaxf(m[r], rm);
        corr[r] = __expf(m[r] - mnew);
        float p0 = __expf(S[0][r] - mnew);
        float p1 = __expf(S[1][r] - mnew);
        S[0][r] = p0; S[1][r] = p1;
        float rs = p0 + p1;
        #pragma unroll
        for (int off = 1; off < 16; off <<= 1)
          rs += __shfl_xor(rs, off, 32);
        l[r] = l[r] * corr[r] + rs;
        m[r] = mnew;
      }
      #pragma unroll
      for (int h = 0; h < 8; ++h)
        #pragma unroll
        for (int r = 0; r < 8; ++r) acc[h][r] *= corr[r];

      #pragma unroll
      for (int nt = 0; nt < 2; ++nt)
        #pragma unroll
        for (int r = 0; r < 8; ++r)
          p_lds[wave][r + 8 * khalf][nt * 16 + col] = (__bf16)S[nt][r];
      asm volatile("s_wait_dscnt 0x0" ::: "memory");
      v16bf pf = load_afrag(&p_lds[wave][col][0], khalf);

      #pragma unroll
      for (int h = 0; h < 8; ++h) {
        v16bf bv = load_bfrag(&v_blkT[h * 16 + col][0], khalf);
        acc[h] = WMMA_BF16(pf, bv, acc[h]);
      }
    }
  }

  #pragma unroll
  for (int h = 0; h < 8; ++h)
    #pragma unroll
    for (int r = 0; r < 8; ++r) {
      int t = t0 + r + 8 * khalf;
      out[(size_t)(b * kT + t) * kH + h * 16 + col] = acc[h][r] / l[r];
    }
}

// ---------------------------------------------------------------------------
extern "C" void kernel_launch(void* const* d_in, const int* in_sizes, int n_in,
                              void* d_out, int out_size, void* d_ws, size_t ws_size,
                              hipStream_t stream) {
  (void)in_sizes; (void)n_in; (void)out_size; (void)ws_size;
  const float* x  = (const float*)d_in[0];
  const float* Wk = (const float*)d_in[1];
  const float* Wq = (const float*)d_in[2];
  const float* Wv = (const float*)d_in[3];
  float* out = (float*)d_out;

  const size_t elems = (size_t)kB * kT * kH;   // 8M elements each
  __bf16* kb  = (__bf16*)d_ws;
  __bf16* qb  = kb + elems;
  __bf16* vbT = qb + elems;

  qkv_bf16_kernel<false><<<dim3((kB * kT) / 128, 2), 256, 0, stream>>>(x, Wk, Wq, kb, qb);
  qkv_bf16_kernel<true ><<<dim3((kB * kT) / 128, 1), 256, 0, stream>>>(x, Wv, Wv, vbT, vbT);
  attn_kernel<<<dim3(kB, kT / 128), 256, 0, stream>>>(kb, qb, vbT, out);
}
